// LLaMABlock_8211977470193
// MI455X (gfx1250) — matrix operand order repaired and checked by reading, not compiled, not run on hardware
//
#include <hip/hip_runtime.h>

// ---------------------------------------------------------------------------
// LLaMA block for MI455X (gfx1250):
//  - all GEMMs via v_wmma_f32_16x16x32_bf16 (f32 accumulate)
//  - LDS double-buffered staging, async global->LDS copies (ASYNCcnt path)
// ---------------------------------------------------------------------------

#define S_LEN 4096
#define D_MODEL 2048
#define D_FF 8192
#define RMS_EPS 1.1920929e-07f

typedef __attribute__((ext_vector_type(16))) __bf16 v16bf;
typedef __attribute__((ext_vector_type(8)))  float  v8f;
typedef __attribute__((ext_vector_type(4)))  int    v4i;

union Frag32B { uint4 u[2]; v16bf v; };

#ifndef __has_builtin
#define __has_builtin(x) 0
#endif

#if __has_builtin(__builtin_amdgcn_global_load_async_to_lds_b128)
#define USE_ASYNC_LDS 1
#else
#define USE_ASYNC_LDS 0
#endif

// AS casts via integer round-trip (low 32 bits of a generic LDS address are
// the LDS byte offset per the aperture rule, so the value is preserved).
#define GLB_V4I(p) ((__attribute__((address_space(1))) v4i*)(size_t)(p))
#define LDS_V4I(p) ((__attribute__((address_space(3))) v4i*)(size_t)(p))

__device__ __forceinline__ void cp16_g2l(const unsigned short* g, unsigned short* l) {
#if USE_ASYNC_LDS
    __builtin_amdgcn_global_load_async_to_lds_b128(GLB_V4I(g), LDS_V4I(l), 0, 0);
#else
    *(uint4*)l = *(const uint4*)g;
#endif
}

__device__ __forceinline__ void async_wait0() {
#if USE_ASYNC_LDS
#if __has_builtin(__builtin_amdgcn_s_wait_asynccnt)
    __builtin_amdgcn_s_wait_asynccnt(0);
#else
    asm volatile("s_wait_asynccnt 0x0" ::: "memory");
#endif
#endif
}

__device__ __forceinline__ unsigned short f2bf(float f) {
    unsigned int u = __float_as_uint(f);
    unsigned int r = (u + 0x7FFFu + ((u >> 16) & 1u)) >> 16;  // RNE
    return (unsigned short)r;
}

__device__ __forceinline__ float wave_max(float v) {
    for (int o = 16; o; o >>= 1) v = fmaxf(v, __shfl_xor(v, o, 32));
    return v;
}
__device__ __forceinline__ float wave_sum(float v) {
    for (int o = 16; o; o >>= 1) v += __shfl_xor(v, o, 32);
    return v;
}

// ---------------------------------------------------------------------------
// GEMM: C[M,N] = alpha * A[M,K](bf16) @ W[N,K](bf16)^T      (x @ W.T form)
// Block: 256 thr = 8 waves. Block tile 64M x 128N, K-step 32, double-buffered
// LDS. Wave (wm 0..3, wn 0..1): 16M x 64N strip = 4 WMMAs / K-step.
// ---------------------------------------------------------------------------
template <bool OUT_BF16>
__global__ __launch_bounds__(256) void gemm_nt_wmma(
    const unsigned short* __restrict__ A, const unsigned short* __restrict__ W,
    void* __restrict__ Cout, int M, int N, int K, float alpha)
{
    __shared__ __align__(16) unsigned short As[2][64 * 32];
    __shared__ __align__(16) unsigned short Bs[2][128 * 32];

    const int tid  = threadIdx.x;
    const int lane = tid & 31;
    const int wave = tid >> 5;
    const int wm   = wave & 3;
    const int wn   = wave >> 2;
    const long m0  = (long)blockIdx.y * 64;
    const long n0  = (long)blockIdx.x * 128;

    v8f acc[4];
#pragma unroll
    for (int j = 0; j < 4; ++j) acc[j] = (v8f){0.f,0.f,0.f,0.f,0.f,0.f,0.f,0.f};

    // staging assignments (all 16B-aligned)
    const int a_row = tid >> 2;               // 64 rows, 4 thr/row
    const int a_col = (tid & 3) << 3;         // 8 ushorts
    const int b_row = tid >> 1;               // 128 rows, 2 thr/row
    const int b_col = (tid & 1) << 4;         // 16 ushorts

    const long a_base = (m0 + a_row) * (long)K + a_col;
    const long w_base = (n0 + b_row) * (long)K + b_col;

    const int am = wm * 16 + (lane & 15);
    const int ak = (lane < 16) ? 0 : 8;       // ISA 16-bit A layout
    const int bk = (lane < 16) ? 0 : 8;       // ISA 16-bit B layout
    const int bn_l = lane & 15;

    // prologue: stage tile 0 into buffer 0
    cp16_g2l(&A[a_base], &As[0][a_row * 32 + a_col]);
    cp16_g2l(&W[w_base], &Bs[0][b_row * 32 + b_col]);
    cp16_g2l(&W[w_base + 8], &Bs[0][b_row * 32 + b_col + 8]);
    async_wait0();
    __syncthreads();

    int buf = 0;
    for (int k0 = 0; k0 < K; k0 += 32) {
        if (k0 + 32 < K) {  // overlap next tile's copy with this tile's WMMAs
            cp16_g2l(&A[a_base + k0 + 32], &As[buf ^ 1][a_row * 32 + a_col]);
            cp16_g2l(&W[w_base + k0 + 32], &Bs[buf ^ 1][b_row * 32 + b_col]);
            cp16_g2l(&W[w_base + k0 + 40], &Bs[buf ^ 1][b_row * 32 + b_col + 8]);
        }

        // preload all fragments of this K-step, then run the WMMA chain
        Frag32B af;
        af.u[0] = *(const uint4*)&As[buf][am * 32 + ak];
        af.u[1] = *(const uint4*)&As[buf][am * 32 + ak + 16];
        Frag32B bfr[4];
#pragma unroll
        for (int j = 0; j < 4; ++j) {
            const int bn = wn * 64 + j * 16 + bn_l;
            bfr[j].u[0] = *(const uint4*)&Bs[buf][bn * 32 + bk];
            bfr[j].u[1] = *(const uint4*)&Bs[buf][bn * 32 + bk + 16];
        }
#pragma unroll
        for (int j = 0; j < 4; ++j)
            acc[j] = __builtin_amdgcn_wmma_f32_16x16x32_bf16(
                false, af.v, false, bfr[j].v, (short)0, acc[j], false, false);

        async_wait0();
        __syncthreads();
        buf ^= 1;
    }

    // C layout: VGPR r, lane: N = lane%16, M = r + (lane<16 ? 0 : 8)
    const int cn  = lane & 15;
    const int cmb = (lane < 16) ? 0 : 8;
#pragma unroll
    for (int j = 0; j < 4; ++j) {
        const long ncol = n0 + wn * 64 + j * 16 + cn;
#pragma unroll
        for (int r = 0; r < 8; ++r) {
            const long row = m0 + wm * 16 + cmb + r;
            const float val = acc[j][r] * alpha;
            if (OUT_BF16)
                ((unsigned short*)Cout)[row * (long)N + ncol] = f2bf(val);
            else
                ((float*)Cout)[row * (long)N + ncol] = val;
        }
    }
}

// ---------------------------------------------------------------------------
// Fused SwiGLU: mid[M,N] = bf16( silu(A@Wg^T) * (A@Wu^T) ), N = d_ff
// Same tiling; shared A fragment feeds two WMMA chains (8 wmma / K-step).
// ---------------------------------------------------------------------------
__global__ __launch_bounds__(256) void gemm_swiglu_wmma(
    const unsigned short* __restrict__ A,
    const unsigned short* __restrict__ Wg,
    const unsigned short* __restrict__ Wu,
    unsigned short* __restrict__ Mid, int M, int N, int K)
{
    __shared__ __align__(16) unsigned short As[2][64 * 32];
    __shared__ __align__(16) unsigned short Bg[2][128 * 32];
    __shared__ __align__(16) unsigned short Bu[2][128 * 32];

    const int tid  = threadIdx.x;
    const int lane = tid & 31;
    const int wave = tid >> 5;
    const int wm   = wave & 3;
    const int wn   = wave >> 2;
    const long m0  = (long)blockIdx.y * 64;
    const long n0  = (long)blockIdx.x * 128;

    v8f accg[4], accu[4];
#pragma unroll
    for (int j = 0; j < 4; ++j) {
        accg[j] = (v8f){0.f,0.f,0.f,0.f,0.f,0.f,0.f,0.f};
        accu[j] = (v8f){0.f,0.f,0.f,0.f,0.f,0.f,0.f,0.f};
    }

    const int a_row = tid >> 2, a_col = (tid & 3) << 3;
    const int b_row = tid >> 1, b_col = (tid & 1) << 4;
    const long a_base = (m0 + a_row) * (long)K + a_col;
    const long w_base = (n0 + b_row) * (long)K + b_col;
    const int am = wm * 16 + (lane & 15);
    const int ak = (lane < 16) ? 0 : 8;
    const int bk = (lane < 16) ? 0 : 8;
    const int bn_l = lane & 15;

    cp16_g2l(&A[a_base], &As[0][a_row * 32 + a_col]);
    cp16_g2l(&Wg[w_base], &Bg[0][b_row * 32 + b_col]);
    cp16_g2l(&Wg[w_base + 8], &Bg[0][b_row * 32 + b_col + 8]);
    cp16_g2l(&Wu[w_base], &Bu[0][b_row * 32 + b_col]);
    cp16_g2l(&Wu[w_base + 8], &Bu[0][b_row * 32 + b_col + 8]);
    async_wait0();
    __syncthreads();

    int buf = 0;
    for (int k0 = 0; k0 < K; k0 += 32) {
        if (k0 + 32 < K) {
            cp16_g2l(&A[a_base + k0 + 32], &As[buf ^ 1][a_row * 32 + a_col]);
            cp16_g2l(&Wg[w_base + k0 + 32], &Bg[buf ^ 1][b_row * 32 + b_col]);
            cp16_g2l(&Wg[w_base + k0 + 40], &Bg[buf ^ 1][b_row * 32 + b_col + 8]);
            cp16_g2l(&Wu[w_base + k0 + 32], &Bu[buf ^ 1][b_row * 32 + b_col]);
            cp16_g2l(&Wu[w_base + k0 + 40], &Bu[buf ^ 1][b_row * 32 + b_col + 8]);
        }

        Frag32B af;
        af.u[0] = *(const uint4*)&As[buf][am * 32 + ak];
        af.u[1] = *(const uint4*)&As[buf][am * 32 + ak + 16];
        Frag32B fg[4], fu[4];
#pragma unroll
        for (int j = 0; j < 4; ++j) {
            const int bn = wn * 64 + j * 16 + bn_l;
            fg[j].u[0] = *(const uint4*)&Bg[buf][bn * 32 + bk];
            fg[j].u[1] = *(const uint4*)&Bg[buf][bn * 32 + bk + 16];
            fu[j].u[0] = *(const uint4*)&Bu[buf][bn * 32 + bk];
            fu[j].u[1] = *(const uint4*)&Bu[buf][bn * 32 + bk + 16];
        }
#pragma unroll
        for (int j = 0; j < 4; ++j) {
            accg[j] = __builtin_amdgcn_wmma_f32_16x16x32_bf16(
                false, af.v, false, fg[j].v, (short)0, accg[j], false, false);
            accu[j] = __builtin_amdgcn_wmma_f32_16x16x32_bf16(
                false, af.v, false, fu[j].v, (short)0, accu[j], false, false);
        }

        async_wait0();
        __syncthreads();
        buf ^= 1;
    }

    const int cn  = lane & 15;
    const int cmb = (lane < 16) ? 0 : 8;
#pragma unroll
    for (int j = 0; j < 4; ++j) {
        const long ncol = n0 + wn * 64 + j * 16 + cn;
#pragma unroll
        for (int r = 0; r < 8; ++r) {
            const long row = m0 + wm * 16 + cmb + r;
            const float g = accg[j][r];
            const float u = accu[j][r];
            const float silu = g * (1.0f / (1.0f + __expf(-g)));
            Mid[row * (long)N + ncol] = f2bf(silu * u);
        }
    }
}

// ---------------------------------------------------------------------------
// Elementwise / reduction kernels
// ---------------------------------------------------------------------------
__global__ __launch_bounds__(256) void rmsnorm_bf16(
    const float* __restrict__ x, const float* __restrict__ w,
    unsigned short* __restrict__ out, int Dd)
{
    const long row = blockIdx.x;
    const float* src = x + row * (long)Dd;
    unsigned short* dst = out + row * (long)Dd;
    __shared__ float red[8];
    const int lane = threadIdx.x & 31, wave = threadIdx.x >> 5;

    float ss = 0.f;
    for (int i = threadIdx.x; i < Dd; i += 256) { float v = src[i]; ss += v * v; }
    ss = wave_sum(ss);
    if (lane == 0) red[wave] = ss;
    __syncthreads();
    float tot = 0.f;
#pragma unroll
    for (int i = 0; i < 8; ++i) tot += red[i];
    const float scale = rsqrtf(tot / (float)Dd + RMS_EPS);
    for (int i = threadIdx.x; i < Dd; i += 256)
        dst[i] = f2bf(src[i] * scale * w[i]);
}

__global__ __launch_bounds__(256) void rope_bf16(
    const float* __restrict__ x, const float* __restrict__ c,
    const float* __restrict__ s, unsigned short* __restrict__ out, int Dd)
{
    const long row = blockIdx.x;
    const int half = Dd >> 1;
    const float* src = x + row * (long)Dd;
    unsigned short* dst = out + row * (long)Dd;
    const float* cr = c + row * (long)half;
    const float* sr = s + row * (long)half;
    for (int j = threadIdx.x; j < half; j += 256) {
        const float x1 = src[j], x2 = src[half + j];
        const float cv = cr[j], sv = sr[j];
        dst[j]        = f2bf(x1 * cv - x2 * sv);
        dst[half + j] = f2bf(x1 * sv + x2 * cv);
    }
}

__global__ __launch_bounds__(256) void softmax_rows_bf16(
    const float* __restrict__ Sc, unsigned short* __restrict__ P, int N)
{
    const long row = blockIdx.x;
    const float* src = Sc + row * (long)N;
    unsigned short* dst = P + row * (long)N;
    __shared__ float redm[8];
    __shared__ float reds[8];
    const int lane = threadIdx.x & 31, wave = threadIdx.x >> 5;

    float m = -3.402823466e38f;
    for (int i = threadIdx.x; i < N; i += 256) m = fmaxf(m, src[i]);
    m = wave_max(m);
    if (lane == 0) redm[wave] = m;
    __syncthreads();
    float bm = redm[0];
#pragma unroll
    for (int i = 1; i < 8; ++i) bm = fmaxf(bm, redm[i]);

    float s = 0.f;
    for (int i = threadIdx.x; i < N; i += 256) s += __expf(src[i] - bm);
    s = wave_sum(s);
    if (lane == 0) reds[wave] = s;
    __syncthreads();
    float bs = 0.f;
#pragma unroll
    for (int i = 0; i < 8; ++i) bs += reds[i];
    const float inv = 1.0f / bs;

    for (int i = threadIdx.x; i < N; i += 256)
        dst[i] = f2bf(__expf(src[i] - bm) * inv);
}

__global__ __launch_bounds__(256) void transpose_bf16(
    const unsigned short* __restrict__ in, unsigned short* __restrict__ out,
    int R, int C)  // in [R,C] -> out [C,R]
{
    __shared__ unsigned short tile[32][33];
    const int bx = blockIdx.x * 32;  // C
    const int by = blockIdx.y * 32;  // R
    const int tx = threadIdx.x & 31;
    const int ty = threadIdx.x >> 5; // 8 rows per pass
#pragma unroll
    for (int i = 0; i < 32; i += 8)
        tile[ty + i][tx] = in[(long)(by + ty + i) * C + bx + tx];
    __syncthreads();
#pragma unroll
    for (int i = 0; i < 32; i += 8)
        out[(long)(bx + ty + i) * R + by + tx] = tile[tx][ty + i];
}

__global__ __launch_bounds__(256) void cvt_f32_bf16(
    const float* __restrict__ in, unsigned short* __restrict__ out, long n)
{
    long i = (long)blockIdx.x * 256 + threadIdx.x;
    const long stride = (long)gridDim.x * 256;
    for (; i < n; i += stride) out[i] = f2bf(in[i]);
}

__global__ __launch_bounds__(256) void add_f32(
    const float* __restrict__ a, const float* __restrict__ b,
    float* __restrict__ o, long n)
{
    long i = (long)blockIdx.x * 256 + threadIdx.x;
    const long stride = (long)gridDim.x * 256;
    for (; i < n; i += stride) o[i] = a[i] + b[i];
}

// ---------------------------------------------------------------------------
// Launcher
// ---------------------------------------------------------------------------
extern "C" void kernel_launch(void* const* d_in, const int* in_sizes, int n_in,
                              void* d_out, int out_size, void* d_ws, size_t ws_size,
                              hipStream_t stream)
{
    (void)in_sizes; (void)n_in; (void)out_size; (void)ws_size;
    const int S = S_LEN, D = D_MODEL, FF = D_FF;

    const float* x      = (const float*)d_in[0];
    const float* w_rn1  = (const float*)d_in[1];
    const float* wq     = (const float*)d_in[2];
    const float* wk     = (const float*)d_in[3];
    const float* wv     = (const float*)d_in[4];
    const float* wo     = (const float*)d_in[5];
    const float* w_rn2  = (const float*)d_in[6];
    const float* w_gate = (const float*)d_in[7];
    const float* w_up   = (const float*)d_in[8];
    const float* w_down = (const float*)d_in[9];
    const float* cosb   = (const float*)d_in[10];
    const float* sinb   = (const float*)d_in[11];
    float* out = (float*)d_out;

    char* ws = (char*)d_ws;
    size_t off = 0;
    auto alloc = [&](size_t bytes) -> void* {
        void* p = ws + off;
        off = (off + bytes + 255) & ~(size_t)255;
        return p;
    };
    typedef unsigned short bf_t;

    bf_t* wq_b = (bf_t*)alloc((size_t)D * D * 2);
    bf_t* wk_b = (bf_t*)alloc((size_t)D * D * 2);
    bf_t* wv_b = (bf_t*)alloc((size_t)D * D * 2);
    bf_t* wo_b = (bf_t*)alloc((size_t)D * D * 2);
    bf_t* wg_b = (bf_t*)alloc((size_t)FF * D * 2);
    bf_t* wu_b = (bf_t*)alloc((size_t)FF * D * 2);
    bf_t* wd_b = (bf_t*)alloc((size_t)D * FF * 2);
    bf_t* h_b  = (bf_t*)alloc((size_t)S * D * 2);
    float* q_f = (float*)alloc((size_t)S * D * 4);
    float* k_f = (float*)alloc((size_t)S * D * 4);
    bf_t* q_b  = (bf_t*)alloc((size_t)S * D * 2);
    bf_t* k_b  = (bf_t*)alloc((size_t)S * D * 2);
    bf_t* v_b  = (bf_t*)alloc((size_t)S * D * 2);
    bf_t* vT_b = (bf_t*)alloc((size_t)S * D * 2);
    float* sc  = (float*)alloc((size_t)S * S * 4);
    bf_t* p_b  = (bf_t*)alloc((size_t)S * S * 2);
    bf_t* av_b = (bf_t*)alloc((size_t)S * D * 2);
    float* o_f = (float*)alloc((size_t)S * D * 4);
    float* x1  = (float*)alloc((size_t)S * D * 4);
    bf_t* h2_b = (bf_t*)alloc((size_t)S * D * 2);
    bf_t* mid_b = (bf_t*)alloc((size_t)S * FF * 2);
    float* ffn_f = (float*)alloc((size_t)S * D * 4);

    const dim3 blk(256);
    auto cvt_grid = [](long n) { long g = (n + 255) / 256; return (unsigned)(g > 4096 ? 4096 : g); };

    // 1) weights -> bf16
    cvt_f32_bf16<<<cvt_grid((long)D * D), blk, 0, stream>>>(wq, wq_b, (long)D * D);
    cvt_f32_bf16<<<cvt_grid((long)D * D), blk, 0, stream>>>(wk, wk_b, (long)D * D);
    cvt_f32_bf16<<<cvt_grid((long)D * D), blk, 0, stream>>>(wv, wv_b, (long)D * D);
    cvt_f32_bf16<<<cvt_grid((long)D * D), blk, 0, stream>>>(wo, wo_b, (long)D * D);
    cvt_f32_bf16<<<cvt_grid((long)FF * D), blk, 0, stream>>>(w_gate, wg_b, (long)FF * D);
    cvt_f32_bf16<<<cvt_grid((long)FF * D), blk, 0, stream>>>(w_up, wu_b, (long)FF * D);
    cvt_f32_bf16<<<cvt_grid((long)D * FF), blk, 0, stream>>>(w_down, wd_b, (long)D * FF);

    // 2) h = rmsnorm(x, w_rn1) -> bf16
    rmsnorm_bf16<<<S, blk, 0, stream>>>(x, w_rn1, h_b, D);

    // 3) q/k/v projections
    gemm_nt_wmma<false><<<dim3(D / 128, S / 64), blk, 0, stream>>>(h_b, wq_b, q_f, S, D, D, 1.0f);
    gemm_nt_wmma<false><<<dim3(D / 128, S / 64), blk, 0, stream>>>(h_b, wk_b, k_f, S, D, D, 1.0f);
    gemm_nt_wmma<true ><<<dim3(D / 128, S / 64), blk, 0, stream>>>(h_b, wv_b, v_b, S, D, D, 1.0f);

    // 4) RoPE -> bf16
    rope_bf16<<<S, blk, 0, stream>>>(q_f, cosb, sinb, q_b, D);
    rope_bf16<<<S, blk, 0, stream>>>(k_f, cosb, sinb, k_b, D);

    // 5) V^T for attn@V
    transpose_bf16<<<dim3(D / 32, S / 32), blk, 0, stream>>>(v_b, vT_b, S, D);

    // 6) scores = q @ k^T / sqrt(D)
    const float inv_scale = 1.0f / sqrtf((float)D);
    gemm_nt_wmma<false><<<dim3(S / 128, S / 64), blk, 0, stream>>>(q_b, k_b, sc, S, S, D, inv_scale);

    // 7) softmax rows -> bf16
    softmax_rows_bf16<<<S, blk, 0, stream>>>(sc, p_b, S);

    // 8) av = P @ V = P @ (V^T)^T
    gemm_nt_wmma<true ><<<dim3(D / 128, S / 64), blk, 0, stream>>>(p_b, vT_b, av_b, S, D, S, 1.0f);

    // 9) o = av @ wo^T ; residual 1
    gemm_nt_wmma<false><<<dim3(D / 128, S / 64), blk, 0, stream>>>(av_b, wo_b, o_f, S, D, D, 1.0f);
    add_f32<<<cvt_grid((long)S * D), blk, 0, stream>>>(x, o_f, x1, (long)S * D);

    // 10) h2 = rmsnorm(x1, w_rn2)
    rmsnorm_bf16<<<S, blk, 0, stream>>>(x1, w_rn2, h2_b, D);

    // 11) fused SwiGLU: mid = silu(h2@wg^T) * (h2@wu^T)  (bf16)
    gemm_swiglu_wmma<<<dim3(FF / 128, S / 64), blk, 0, stream>>>(h2_b, wg_b, wu_b, mid_b, S, FF, D);

    // 12) down proj + residual 2 -> out
    gemm_nt_wmma<false><<<dim3(D / 128, S / 64), blk, 0, stream>>>(mid_b, wd_b, ffn_f, S, D, FF, 1.0f);
    add_f32<<<cvt_grid((long)S * D), blk, 0, stream>>>(x1, ffn_f, out, (long)S * D);
}
